// GINPTEmbedder_29025388986839
// MI455X (gfx1250) — compile-verified, matrix operand order checked by reading, not run on hardware
//
#include <hip/hip_runtime.h>
#include <hip/hip_bf16.h>

// ---------------------------------------------------------------------------
// GIN forward for MI455X (gfx1250, wave32, WMMA).
//   - h / activations bf16 (157MB -> L2 resident on 192MB L2)
//   - MLP GEMMs via v_wmma_f32_16x16x32_bf16, f32 accumulate
//   - double-buffered LDS software pipeline: next tile's global loads issue
//     before current tile's WMMAs (one barrier per K-step)
//   - message-passing scatter via global_atomic_add_f32
// ---------------------------------------------------------------------------

typedef __attribute__((ext_vector_type(16))) __bf16 v16bf;
typedef __attribute__((ext_vector_type(8)))  __bf16 v8bf;
typedef __attribute__((ext_vector_type(8)))  float  v8f;

#define HDIM 300
#define H2   600
#define NLAYERS 5
#define BN_EPS 1e-5f

// padded geometry
#define KP1 320   // GEMM1 K (agg channels padded)         multiple of 32
#define NP1 640   // GEMM1 NC padded (X1 cols, Bt1 rows)   multiple of 64
#define KP2 608   // GEMM2 K (X1 cols consumed)            multiple of 32
#define NP2 320   // GEMM2 NC padded (X2 cols, Bt2 rows)   multiple of 64
#define AGS 320   // agg / X2 row stride (f32)
#define X1S 640   // X1 row stride (bf16)

// GEMM tiling: block = 256 threads = 8 waves; tile 128(M) x 64(N) x 32(K).
// Waves arranged 4(M) x 2(N); each wave owns 32x32 = 2x2 WMMA accumulators.
#define BM 128
#define BN 64
#define BK 32
#define LDT 48   // LDS k-stride (elems): 96B rows -> 32B-aligned v16bf loads

// ---------------------------------------------------------------------------
// h0[n][c] = node_emb0[atom[n]][c] + node_emb1[chir[n]][c]   (bf16 out)
// ---------------------------------------------------------------------------
__global__ __launch_bounds__(256)
void k_init_h(const int* __restrict__ atom, const int* __restrict__ chir,
              const float* __restrict__ e0, const float* __restrict__ e1,
              __bf16* __restrict__ h, int n_nodes) {
  long idx = (long)blockIdx.x * blockDim.x + threadIdx.x;
  long total = (long)n_nodes * HDIM;
  if (idx >= total) return;
  int n = (int)(idx / HDIM);
  int c = (int)(idx % HDIM);
  float v = e0[(long)atom[n] * HDIM + c] + e1[(long)chir[n] * HDIM + c];
  h[idx] = (__bf16)v;
}

// ---------------------------------------------------------------------------
// Weight prep: Bt[l][n][k] = bf16(W[l][k][n]), zero-padded buffer (memset'd).
// ---------------------------------------------------------------------------
__global__ __launch_bounds__(256)
void k_prep_w(const float* __restrict__ W, __bf16* __restrict__ Bt,
              int K, int N, int NProws, int KP) {
  long idx = (long)blockIdx.x * blockDim.x + threadIdx.x;
  long per = (long)K * N;
  long total = (long)NLAYERS * per;
  if (idx >= total) return;
  int l = (int)(idx / per);
  long r = idx % per;
  int k = (int)(r / N);
  int n = (int)(r % N);
  Bt[((long)l * NProws + n) * KP + k] = (__bf16)W[idx];
}

__global__ __launch_bounds__(256)
void k_prep_bias(const float* __restrict__ b, float* __restrict__ bp,
                 int N, int NP) {
  long idx = (long)blockIdx.x * blockDim.x + threadIdx.x;
  long total = (long)NLAYERS * NP;
  if (idx >= total) return;
  int l = (int)(idx / NP);
  int n = (int)(idx % NP);
  bp[idx] = (n < N) ? b[(long)l * N + n] : 0.0f;
}

// ---------------------------------------------------------------------------
// agg[dst[e]][c] += (float)h[src[e]][c] + ee0[bond[e]][c] + ee1[dir[e]][c]
// One thread handles 4 consecutive channels of one edge.  agg stride = AGS.
// ---------------------------------------------------------------------------
__global__ __launch_bounds__(256)
void k_scatter(const __bf16* __restrict__ h,
               const float* __restrict__ ee0, const float* __restrict__ ee1,
               const int* __restrict__ bond, const int* __restrict__ dir,
               const int* __restrict__ src, const int* __restrict__ dst,
               float* __restrict__ agg, int n_edges) {
  long idx = (long)blockIdx.x * blockDim.x + threadIdx.x;
  long total = (long)n_edges * (HDIM / 4);
  if (idx >= total) return;
  int e  = (int)(idx / (HDIM / 4));
  int c0 = (int)(idx % (HDIM / 4)) * 4;
  int s = src[e], d = dst[e], b = bond[e], dr = dir[e];
  const __bf16* hs = h + (long)s * HDIM + c0;
  const float* t0 = ee0 + (long)b * HDIM + c0;
  const float* t1 = ee1 + (long)dr * HDIM + c0;
  float* ad = agg + (long)d * AGS + c0;
#pragma unroll
  for (int i = 0; i < 4; ++i) {
    float v = (float)hs[i] + t0[i] + t1[i];
    atomicAdd(&ad[i], v);
  }
}

// ---------------------------------------------------------------------------
// C = act(A @ Bt^T + bias).  All operands pre-padded: staging is branch-free.
//   A : [M, Astride]   f32 (A_F32) or bf16
//   Bt: [*, KP]        bf16, n-major, K-contiguous, zero-padded
//   C : [M, Cstride]   bf16 (OUT_BF16) or f32; all cols in grid range valid
// Double-buffered LDS pipeline: loads for tile k+1 issue before WMMA(tile k).
// ---------------------------------------------------------------------------
template <bool A_F32, bool RELU, bool OUT_BF16>
__global__ __launch_bounds__(256)
void k_gemm(const void* __restrict__ Av, int Astride,
            const __bf16* __restrict__ Bt, int KP,
            const float* __restrict__ bias, void* __restrict__ Cv,
            int Cstride, int M, int Kloop) {
  __shared__ __attribute__((aligned(32))) __bf16 As[2][BM * LDT];
  __shared__ __attribute__((aligned(32))) __bf16 Bs[2][BN * LDT];

  const int tid  = threadIdx.x;
  const int lane = tid & 31;
  const int wave = tid >> 5;
  const int wm   = wave & 3;   // wave tile row (0..3) * 32
  const int wn   = wave >> 2;  // wave tile col (0..1) * 32
  const int m0   = blockIdx.y * BM;
  const int n0   = blockIdx.x * BN;

  v8f acc[2][2] = {};

  const int lr = lane & 15;          // row (A) / col (B) within 16
  const int kb = (lane >> 4) * 16;   // K sub-range base for this lane

  // staging coordinates (branch-free; rows clamped, pads are zero)
  const int arow = tid >> 1;                    // 0..127
  const int aks  = (tid & 1) * 16;              // 0 or 16
  const int agr  = min(m0 + arow, M - 1);       // clamp, discard at store
  const int brow = tid >> 2;                    // 0..63
  const int bks  = (tid & 3) * 8;               // 0,8,16,24

  // issue global loads for one K-tile into registers
  float4 fr[4];
  v16bf  abf;
  v8bf   bb;
  auto stage_load = [&](int k0) {
    if constexpr (A_F32) {
      const float4* srcp =
          (const float4*)((const float*)Av + (long)agr * Astride + k0 + aks);
#pragma unroll
      for (int t = 0; t < 4; ++t) fr[t] = srcp[t];
    } else {
      abf = *(const v16bf*)((const __bf16*)Av + (long)agr * Astride + k0 + aks);
    }
    bb = *(const v8bf*)(Bt + (long)(n0 + brow) * KP + k0 + bks);
  };
  // commit staged registers to LDS buffer `buf`
  auto stage_store = [&](int buf) {
    if constexpr (A_F32) {
      v16bf o;
#pragma unroll
      for (int t = 0; t < 4; ++t) {
        o[4 * t + 0] = (__bf16)fr[t].x;
        o[4 * t + 1] = (__bf16)fr[t].y;
        o[4 * t + 2] = (__bf16)fr[t].z;
        o[4 * t + 3] = (__bf16)fr[t].w;
      }
      *(v16bf*)&As[buf][arow * LDT + aks] = o;
    } else {
      *(v16bf*)&As[buf][arow * LDT + aks] = abf;
    }
    *(v8bf*)&Bs[buf][brow * LDT + bks] = bb;
  };
  // 4 WMMAs from LDS buffer `buf`
  auto compute = [&](int buf) {
    v16bf a0 = *(const v16bf*)&As[buf][(wm * 32 + 0 + lr) * LDT + kb];
    v16bf a1 = *(const v16bf*)&As[buf][(wm * 32 + 16 + lr) * LDT + kb];
    v16bf b0 = *(const v16bf*)&Bs[buf][(wn * 32 + 0 + lr) * LDT + kb];
    v16bf b1 = *(const v16bf*)&Bs[buf][(wn * 32 + 16 + lr) * LDT + kb];
    acc[0][0] = __builtin_amdgcn_wmma_f32_16x16x32_bf16(
        false, a0, false, b0, (short)0, acc[0][0], false, false);
    acc[0][1] = __builtin_amdgcn_wmma_f32_16x16x32_bf16(
        false, a0, false, b1, (short)0, acc[0][1], false, false);
    acc[1][0] = __builtin_amdgcn_wmma_f32_16x16x32_bf16(
        false, a1, false, b0, (short)0, acc[1][0], false, false);
    acc[1][1] = __builtin_amdgcn_wmma_f32_16x16x32_bf16(
        false, a1, false, b1, (short)0, acc[1][1], false, false);
  };

  // prologue: fill buffer 0
  stage_load(0);
  stage_store(0);
  __syncthreads();

  int cur = 0;
  for (int k0 = BK; k0 < Kloop; k0 += BK) {
    stage_load(k0);        // global loads for next tile in flight
    compute(cur);          // WMMA on current tile hides load latency
    stage_store(1 - cur);  // safe: 1-cur last read before previous barrier
    __syncthreads();
    cur ^= 1;
  }
  compute(cur);

  // --- epilogue: D layout => lane: col = lane&15, rows = (lane>>4)*8 + r ---
  const int rhalf = (lane >> 4) * 8;
#pragma unroll
  for (int j = 0; j < 2; ++j) {
    int gcol = n0 + wn * 32 + j * 16 + lr;
    float bv = bias[gcol];  // pre-padded, unguarded
#pragma unroll
    for (int i = 0; i < 2; ++i) {
#pragma unroll
      for (int r = 0; r < 8; ++r) {
        int grow = m0 + wm * 32 + i * 16 + rhalf + r;
        if (grow >= M) continue;
        float v = acc[i][j][r] + bv;
        if (RELU) v = fmaxf(v, 0.0f);
        if (OUT_BF16)
          ((__bf16*)Cv)[(long)grow * Cstride + gcol] = (__bf16)v;
        else
          ((float*)Cv)[(long)grow * Cstride + gcol] = v;
      }
    }
  }
}

// ---------------------------------------------------------------------------
// BN stats: per-channel sum / sum-of-squares over nodes (x stride = AGS).
// ---------------------------------------------------------------------------
__global__ __launch_bounds__(256)
void k_bn_stats(const float* __restrict__ x, float* __restrict__ stats,
                int n_nodes) {
  int r0 = blockIdx.x * 128;
  int rend = min(r0 + 128, n_nodes);
  for (int c = threadIdx.x; c < HDIM; c += 256) {
    float s = 0.0f, q = 0.0f;
    for (int r = r0; r < rend; ++r) {
      float v = x[(long)r * AGS + c];
      s += v;
      q += v * v;
    }
    atomicAdd(&stats[c], s);
    atomicAdd(&stats[HDIM + c], q);
  }
}

// ---------------------------------------------------------------------------
// BN normalize. LAST=false: h = bf16(relu(gamma*(x-mu)*rsqrt(var+eps)+beta)).
// LAST=true: pooled-sum atomics: gsum[seg[n]][c] += norm (no relu).
// ---------------------------------------------------------------------------
template <bool LAST>
__global__ __launch_bounds__(256)
void k_bn_norm(const float* __restrict__ x, const float* __restrict__ stats,
               const float* __restrict__ gamma, const float* __restrict__ beta,
               __bf16* __restrict__ h, const int* __restrict__ seg,
               float* __restrict__ gsum, int n_nodes, float invN) {
  long idx = (long)blockIdx.x * blockDim.x + threadIdx.x;
  long total = (long)n_nodes * HDIM;
  if (idx >= total) return;
  int n = (int)(idx / HDIM);
  int c = (int)(idx % HDIM);
  float mu  = stats[c] * invN;
  float var = stats[HDIM + c] * invN - mu * mu;
  float v = gamma[c] * (x[(long)n * AGS + c] - mu) * rsqrtf(var + BN_EPS) +
            beta[c];
  if (!LAST) {
    h[idx] = (__bf16)fmaxf(v, 0.0f);
  } else {
    atomicAdd(&gsum[(long)seg[n] * HDIM + c], v);
  }
}

__global__ __launch_bounds__(256)
void k_counts(const int* __restrict__ seg, float* __restrict__ cnt,
              int n_nodes) {
  int n = blockIdx.x * blockDim.x + threadIdx.x;
  if (n >= n_nodes) return;
  atomicAdd(&cnt[seg[n]], 1.0f);
}

__global__ __launch_bounds__(256)
void k_pool(const float* __restrict__ gsum, const float* __restrict__ cnt,
            float* __restrict__ out, int n_graphs) {
  long idx = (long)blockIdx.x * blockDim.x + threadIdx.x;
  long total = (long)n_graphs * HDIM;
  if (idx >= total) return;
  int g = (int)(idx / HDIM);
  float v = gsum[idx] / fmaxf(cnt[g], 1.0f);
  out[idx] = fmaxf(v, 0.0f);
}

// ---------------------------------------------------------------------------
static inline size_t align256(size_t x) { return (x + 255) & ~(size_t)255; }

extern "C" void kernel_launch(void* const* d_in, const int* in_sizes, int n_in,
                              void* d_out, int out_size, void* d_ws,
                              size_t ws_size, hipStream_t stream) {
  const int* atom = (const int*)d_in[0];
  const int* chir = (const int*)d_in[1];
  const int* bond = (const int*)d_in[2];
  const int* dir  = (const int*)d_in[3];
  const int* src  = (const int*)d_in[4];
  const int* dst  = (const int*)d_in[5];
  const int* seg  = (const int*)d_in[6];
  // d_in[7] = n_graphs scalar (device); derive host-side from out_size
  const float* ne0   = (const float*)d_in[8];
  const float* ne1   = (const float*)d_in[9];
  const float* ee0   = (const float*)d_in[10];  // [5,6,300]
  const float* ee1   = (const float*)d_in[11];  // [5,3,300]
  const float* W1    = (const float*)d_in[12];  // [5,300,600]
  const float* b1    = (const float*)d_in[13];  // [5,600]
  const float* W2    = (const float*)d_in[14];  // [5,600,300]
  const float* b2    = (const float*)d_in[15];  // [5,300]
  const float* gamma = (const float*)d_in[16];  // [5,300]
  const float* beta  = (const float*)d_in[17];  // [5,300]

  const int Nn = in_sizes[0];          // 262144
  const int Ne = in_sizes[2];          // 524288
  const int G  = out_size / HDIM;      // 8192
  float* out = (float*)d_out;

  // workspace carve-up (X2 aliases agg: agg dead once GEMM1 consumed it)
  char* p = (char*)d_ws;
  __bf16* h = (__bf16*)p;   p += align256((size_t)Nn * HDIM * 2);
  float* agg = (float*)p;   p += align256((size_t)Nn * AGS * 4);
  __bf16* X1 = (__bf16*)p;  p += align256((size_t)Nn * X1S * 2);
  __bf16* Bt1 = (__bf16*)p; p += align256((size_t)NLAYERS * NP1 * KP1 * 2);
  __bf16* Bt2 = (__bf16*)p; p += align256((size_t)NLAYERS * NP2 * KP2 * 2);
  float* b1p = (float*)p;   p += align256((size_t)NLAYERS * NP1 * 4);
  float* b2p = (float*)p;   p += align256((size_t)NLAYERS * NP2 * 4);
  float* stats = (float*)p; p += align256((size_t)2 * HDIM * 4);
  float* gsum = (float*)p;  p += align256((size_t)G * HDIM * 4);
  float* cnt = (float*)p;   p += align256((size_t)G * 4);
  (void)ws_size; (void)n_in;
  float* X2 = agg;  // alias

  const long NH = (long)Nn * HDIM;
  const int blkNH = (int)((NH + 255) / 256);
  const int gemmMy = (Nn + BM - 1) / BM;

  // one-time (per launch) prep: padded bf16 transposed weights, padded biases
  hipMemsetAsync(Bt1, 0, (size_t)NLAYERS * NP1 * KP1 * 2, stream);
  hipMemsetAsync(Bt2, 0, (size_t)NLAYERS * NP2 * KP2 * 2, stream);
  {
    long t1 = (long)NLAYERS * HDIM * H2;
    k_prep_w<<<(int)((t1 + 255) / 256), 256, 0, stream>>>(W1, Bt1, HDIM, H2,
                                                          NP1, KP1);
    k_prep_w<<<(int)((t1 + 255) / 256), 256, 0, stream>>>(W2, Bt2, H2, HDIM,
                                                          NP2, KP2);
    long tb1 = (long)NLAYERS * NP1, tb2 = (long)NLAYERS * NP2;
    k_prep_bias<<<(int)((tb1 + 255) / 256), 256, 0, stream>>>(b1, b1p, H2, NP1);
    k_prep_bias<<<(int)((tb2 + 255) / 256), 256, 0, stream>>>(b2, b2p, HDIM,
                                                              NP2);
  }

  hipMemsetAsync(gsum, 0, (size_t)G * HDIM * 4, stream);
  hipMemsetAsync(cnt, 0, (size_t)G * 4, stream);
  k_counts<<<(Nn + 255) / 256, 256, 0, stream>>>(seg, cnt, Nn);
  k_init_h<<<blkNH, 256, 0, stream>>>(atom, chir, ne0, ne1, h, Nn);

  for (int l = 0; l < NLAYERS; ++l) {
    const float* ee0l = ee0 + (long)l * 6 * HDIM;
    const float* ee1l = ee1 + (long)l * 3 * HDIM;

    hipMemsetAsync(agg, 0, (size_t)Nn * AGS * 4, stream);
    long st = (long)Ne * (HDIM / 4);
    k_scatter<<<(int)((st + 255) / 256), 256, 0, stream>>>(
        h, ee0l, ee1l, bond, dir, src, dst, agg, Ne);

    // X1 = relu(agg @ W1 + b1)   [N,320]@[320,640] -> bf16 (pads exact zero)
    k_gemm<true, true, true><<<dim3(NP1 / BN, gemmMy), 256, 0, stream>>>(
        agg, AGS, Bt1 + (long)l * NP1 * KP1, KP1, b1p + (long)l * NP1, X1, X1S,
        Nn, KP1);
    // X2 = X1 @ W2 + b2          [N,608]@[608,320] -> f32 (aliases agg)
    k_gemm<false, false, false><<<dim3(NP2 / BN, gemmMy), 256, 0, stream>>>(
        X1, X1S, Bt2 + (long)l * NP2 * KP2, KP2, b2p + (long)l * NP2, X2, AGS,
        Nn, KP2);

    hipMemsetAsync(stats, 0, (size_t)2 * HDIM * 4, stream);
    k_bn_stats<<<(Nn + 127) / 128, 256, 0, stream>>>(X2, stats, Nn);

    float invN = 1.0f / (float)Nn;
    if (l < NLAYERS - 1) {
      k_bn_norm<false><<<blkNH, 256, 0, stream>>>(
          X2, stats, gamma + (long)l * HDIM, beta + (long)l * HDIM, h, nullptr,
          nullptr, Nn, invN);
    } else {
      k_bn_norm<true><<<blkNH, 256, 0, stream>>>(
          X2, stats, gamma + (long)l * HDIM, beta + (long)l * HDIM, nullptr,
          seg, gsum, Nn, invN);
    }
  }

  long pt = (long)G * HDIM;
  k_pool<<<(int)((pt + 255) / 256), 256, 0, stream>>>(gsum, cnt, out, G);
}
